// Generator_22196390985916
// MI455X (gfx1250) — matrix operand order repaired and checked by reading, not compiled, not run on hardware
//
#include <hip/hip_runtime.h>
#include <hip/hip_bf16.h>

typedef __attribute__((ext_vector_type(16))) _Float16 v16h;
typedef __attribute__((ext_vector_type(8)))  float    v8f;

#define B_   8
#define C_   64
#define H_   128
#define W_   128
#define HW_  16384
#define F_   5
#define L_   512
#define HC_  128     // spade hidden channels
#define NSTEP 36     // K = 9*128 = 1152 -> 36 steps of 32

// ---------------------------------------------------------------- region id
__global__ void k_rid(const float* __restrict__ seg, int* __restrict__ rid) {
    int t = blockIdx.x * 256 + threadIdx.x;          // B*HW = 131072
    int b = t / HW_, hw = t % HW_;
    float best = -1.f; int bj = 0;
    #pragma unroll
    for (int j = 0; j < F_; ++j) {
        float v = seg[((size_t)(b * F_ + j)) * HW_ + hw];
        if (v > best) { best = v; bj = j; }
    }
    rid[t] = bj;
}

// ---------------------------------------------------------------- instance norm
__global__ void __launch_bounds__(256) k_instnorm(const float* __restrict__ x,
                                                  float* __restrict__ xn) {
    int bc = blockIdx.x;                              // B*C = 512 blocks
    const float* p = x + (size_t)bc * HW_;
    float s = 0.f, s2 = 0.f;
    for (int i = threadIdx.x; i < HW_; i += 256) { float v = p[i]; s += v; s2 += v * v; }
    __shared__ float rs[256], rq[256];
    rs[threadIdx.x] = s; rq[threadIdx.x] = s2; __syncthreads();
    for (int o = 128; o > 0; o >>= 1) {
        if (threadIdx.x < o) { rs[threadIdx.x] += rs[threadIdx.x + o];
                               rq[threadIdx.x] += rq[threadIdx.x + o]; }
        __syncthreads();
    }
    float mu  = rs[0] * (1.f / HW_);
    float var = rq[0] * (1.f / HW_) - mu * mu;
    float inv = rsqrtf(var + 1e-5f);
    float* q = xn + (size_t)bc * HW_;
    for (int i = threadIdx.x; i < HW_; i += 256) q[i] = (p[i] - mu) * inv;
}

// ---------------------------------------------------------------- per-region FC
__global__ void k_fc(const float* __restrict__ codes, const float* __restrict__ Wfc,
                     const float* __restrict__ bfc, float* __restrict__ mu) {
    int t = blockIdx.x * 256 + threadIdx.x;          // B*F*L = 20480
    int m = t & (L_ - 1); int j = (t >> 9) % F_; int b = t / (F_ * L_);
    const float* c = codes + ((size_t)b * F_ + j) * L_;
    const float* w = Wfc + ((size_t)j * L_ + m) * L_;
    float a = bfc[j * L_ + m];
    for (int l = 0; l < L_; ++l) a += c[l] * w[l];
    mu[t] = fmaxf(a, 0.f);
}

// ------------------------------------------- region conv collapsed to tables
// T[b,j,tap,o] = sum_m conv_w[o,m,tap] * middle_mu[b,j,m]
__global__ void k_regionT(const float* __restrict__ cgw, const float* __restrict__ cbw,
                          const float* __restrict__ mu,
                          float* __restrict__ Tg, float* __restrict__ Tb) {
    int t = blockIdx.x * 256 + threadIdx.x;          // 2*B*F*9*64 = 46080
    int o = t & 63; int tap = (t >> 6) % 9; int j = (t / (64 * 9)) % F_;
    int rem = t / (64 * 9 * F_); int b = rem % B_; int mat = rem / B_;
    const float* w  = (mat ? cbw : cgw) + (size_t)o * L_ * 9 + tap;
    const float* mv = mu + ((size_t)b * F_ + j) * L_;
    float a = 0.f;
    for (int l = 0; l < L_; ++l) a += w[(size_t)l * 9] * mv[l];
    float* T = mat ? Tb : Tg;
    T[(((size_t)(b * F_ + j)) * 9 + tap) * 64 + o] = a;
}

// --------------------------------- pre-swizzle spade weights into B fragments
// frag[mat][s][nt][lane][h], B 32x16 f16 frags in this kernel's contiguous k order (as A): lanes 0-15: K=h, lanes 16-31: K=16+h
__global__ void k_prep(const float* __restrict__ gw, const float* __restrict__ bw,
                       _Float16* __restrict__ frag) {
    int t = blockIdx.x * 256 + threadIdx.x;          // 2*36*4*32*16 = 147456
    int h = t & 15; int lane = (t >> 4) & 31; int nt = (t >> 9) & 3;
    int s = (t >> 11) % NSTEP; int mat = t / (NSTEP * 2048);
    int K  = (lane < 16) ? h : 16 + h;
    int kg = s * 32 + K;
    int tap = kg >> 7; int c = kg & 127;
    int o = nt * 16 + (lane & 15);
    const float* w = mat ? bw : gw;                  // [O=64][IC=128][3][3]
    frag[t] = (_Float16)w[((size_t)o * HC_ + c) * 9 + tap];
}

// ---------------------------------------------------------------- spade conv1 (3->128)
__global__ void __launch_bounds__(256) k_conv1(const float* __restrict__ mask,
                                               const float* __restrict__ sw,
                                               const float* __restrict__ sb,
                                               _Float16* __restrict__ actv) {
    __shared__ float mrow[3][3][132];
    int blk = blockIdx.x;                             // 8*64*128 = 65536
    int y = blk & 127; int ocg = (blk >> 7) & 63; int b = blk >> 13;
    int tid = threadIdx.x;
    for (int e = tid; e < 3 * 3 * 130; e += 256) {
        int ic = e / 390; int rr = (e / 130) % 3; int xi = e % 130;
        int yy = y + rr - 1; int xx = xi - 1;
        float v = 0.f;
        if (yy >= 0 && yy < H_ && xx >= 0 && xx < W_)
            v = mask[((size_t)(b * 3 + ic)) * HW_ + yy * W_ + xx];
        mrow[ic][rr][xi] = v;
    }
    __syncthreads();
    int oc = ocg * 2 + (tid >> 7); int x = tid & 127;
    float acc = sb[oc];
    #pragma unroll
    for (int ic = 0; ic < 3; ++ic)
        #pragma unroll
        for (int ky = 0; ky < 3; ++ky)
            #pragma unroll
            for (int kx = 0; kx < 3; ++kx)
                acc += sw[((oc * 3 + ic) * 9) + ky * 3 + kx] * mrow[ic][ky][x + kx];
    actv[((size_t)(b * HC_ + oc) * H_ + y) * W_ + x] = (_Float16)fmaxf(acc, 0.f);
}

// ----------------- main fused kernel: WMMA spade convs + region gather + apply
__global__ void __launch_bounds__(256) k_main(
    const _Float16* __restrict__ actv, const _Float16* __restrict__ frag,
    const float* __restrict__ sg_b, const float* __restrict__ sb_b,
    const int* __restrict__ rid,
    const float* __restrict__ Tg, const float* __restrict__ Tb,
    const float* __restrict__ cg_b, const float* __restrict__ cb_b,
    const float* __restrict__ xnorm,
    const float* __restrict__ blend_g, const float* __restrict__ blend_b,
    float* __restrict__ out) {
    // Ach padded to 66 rows: logical pixel m lives in row m+1, rows 0/65 are
    // scratch so the dx-shifted staging stores are fully unconditional.
    __shared__ __align__(16) _Float16 Ach[66][32];   // 4.1 KB act chunk [m+1][kk]
    __shared__ float Ssp[2][64][64];                 // 32 KB gamma/beta spade
    int blk = blockIdx.x;                            // 8*128*2 = 2048
    int xt = blk & 1; int y = (blk >> 1) & 127; int b = blk >> 8;
    int x0 = xt * 64;
    int tid = threadIdx.x, lane = tid & 31, wave = tid >> 5;
    int mt = wave & 3, mat = wave >> 2;              // mat 0=gamma, 1=beta
    int m_base = mt * 16;

    v8f acc[4]; acc[0] = (v8f){}; acc[1] = (v8f){}; acc[2] = (v8f){}; acc[3] = (v8f){};
    int ml   = (lane < 16) ? lane : lane - 16;
    int kklo = (lane < 16) ? 0 : 16;
    int mrow = m_base + ml + 1;                      // +1: padded row shift
    int kk = tid & 31, mg = tid >> 5;
    const _Float16* chan = actv + (size_t)(b * HC_ + kk) * HW_;  // per-thread channel base
    const _Float16* fb0  = frag + (size_t)mat * NSTEP * 4 * 512 + (size_t)lane * 16;

    for (int tap = 0; tap < 9; ++tap) {
        int dy = tap / 3 - 1, dx = tap % 3 - 1;      // 9x only
        int yy = y + dy;
        bool yok = (yy >= 0) && (yy < H_);
        const _Float16* rowp = chan + (size_t)(yok ? yy : 0) * W_;   // + c*HW_ later
        int m_edge = (dx == 1) ? 63 : 0;
        int x_edge = x0 + dx + m_edge;               // x0+64 or x0-1
        bool do_edge = (mg == 0) && (dx != 0);
        #pragma unroll
        for (int cb = 0; cb < 4; ++cb) {
            int s = tap * 4 + cb;
            __syncthreads();                          // chunk reuse barrier
            {
                // Aligned 16B chunk at x0+mg*8 (always in-bounds in x); the dx
                // shift is absorbed into the (padded) LDS store row.
                const _Float16* src = rowp + (size_t)(cb * 32) * HW_;
                uint4 d = make_uint4(0u, 0u, 0u, 0u);
                if (yok) d = *(const uint4*)(src + x0 + mg * 8);
                const _Float16* hp = (const _Float16*)&d;
                #pragma unroll
                for (int i = 0; i < 8; ++i)
                    Ach[mg * 8 + i - dx + 1][kk] = hp[i];     // unconditional
                if (do_edge) {                        // single boundary column
                    _Float16 v = (_Float16)0.f;
                    if (yok && x_edge >= 0 && x_edge < W_) v = src[x_edge];
                    Ach[m_edge + 1][kk] = v;
                }
            }
            // preload all 4 B fragments (global, L2-hot) before LDS reads + WMMAs
            union { uint4 u[2]; v16h v; } bf[4];
            const uint4* gp = (const uint4*)(fb0 + (size_t)s * 4 * 512);
            #pragma unroll
            for (int nt = 0; nt < 4; ++nt) { bf[nt].u[0] = gp[nt * 64];
                                             bf[nt].u[1] = gp[nt * 64 + 1]; }
            __syncthreads();
            union { uint4 u[2]; v16h v; } af;
            af.u[0] = *(const uint4*)&Ach[mrow][kklo];
            af.u[1] = *(const uint4*)&Ach[mrow][kklo + 8];
            #pragma unroll
            for (int nt = 0; nt < 4; ++nt)
                acc[nt] = __builtin_amdgcn_wmma_f32_16x16x32_f16(
                    false, af.v, false, bf[nt].v, (short)0, acc[nt], false, false);
        }
    }

    // deposit spade results (+conv bias) into LDS
    const float* sbias = (mat == 0) ? sg_b : sb_b;
    #pragma unroll
    for (int nt = 0; nt < 4; ++nt)
        #pragma unroll
        for (int r = 0; r < 8; ++r) {
            int m_local = (lane < 16) ? r : 8 + r;
            int o = nt * 16 + (lane & 15);
            Ssp[mat][m_base + m_local][o] = acc[nt][r] + sbias[o];
        }
    __syncthreads();

    // fused epilogue: region-table gather + blend + apply to x_norm
    float ga = 1.f / (1.f + expf(-blend_g[0]));
    float ba = 1.f / (1.f + expf(-blend_b[0]));
    int m = tid & 63, og = tid >> 6;
    int xx = x0 + m;
    float gsum[16], bsum[16];
    #pragma unroll
    for (int oo = 0; oo < 16; ++oo) { int o = og * 16 + oo; gsum[oo] = cg_b[o]; bsum[oo] = cb_b[o]; }
    for (int tap = 0; tap < 9; ++tap) {
        int dy = tap / 3 - 1, dx = tap % 3 - 1;
        int yy = y + dy, x2 = xx + dx;
        if (yy < 0 || yy >= H_ || x2 < 0 || x2 >= W_) continue;  // SAME zero-pad
        int j = rid[b * HW_ + yy * W_ + x2];
        const float* tg = Tg + (((size_t)(b * F_ + j)) * 9 + tap) * 64 + og * 16;
        const float* tb = Tb + (((size_t)(b * F_ + j)) * 9 + tap) * 64 + og * 16;
        #pragma unroll
        for (int oo = 0; oo < 16; ++oo) { gsum[oo] += tg[oo]; bsum[oo] += tb[oo]; }
    }
    #pragma unroll
    for (int oo = 0; oo < 16; ++oo) {
        int o = og * 16 + oo;
        float gf = ga * gsum[oo] + (1.f - ga) * Ssp[0][m][o];
        float bf2 = ba * bsum[oo] + (1.f - ba) * Ssp[1][m][o];
        size_t idx = ((size_t)(b * C_ + o) * H_ + y) * W_ + xx;
        out[idx] = xnorm[idx] * (1.f + gf) + bf2;
    }
}

// ---------------------------------------------------------------- launcher
extern "C" void kernel_launch(void* const* d_in, const int* in_sizes, int n_in,
                              void* d_out, int out_size, void* d_ws, size_t ws_size,
                              hipStream_t stream) {
    const float* x       = (const float*)d_in[0];
    const float* segmap  = (const float*)d_in[1];
    const float* codes   = (const float*)d_in[2];
    const float* mask    = (const float*)d_in[3];
    const float* W_fc    = (const float*)d_in[4];
    const float* b_fc    = (const float*)d_in[5];
    const float* cgw     = (const float*)d_in[6];
    const float* cgb     = (const float*)d_in[7];
    const float* cbw     = (const float*)d_in[8];
    const float* cbb     = (const float*)d_in[9];
    const float* ssw     = (const float*)d_in[10];
    const float* ssb     = (const float*)d_in[11];
    const float* sgw     = (const float*)d_in[12];
    const float* sgb     = (const float*)d_in[13];
    const float* sbw     = (const float*)d_in[14];
    const float* sbb     = (const float*)d_in[15];
    const float* blend_g = (const float*)d_in[16];
    const float* blend_b = (const float*)d_in[17];
    float* out = (float*)d_out;

    char* ws = (char*)d_ws;
    size_t off = 0;
    float*    xnorm = (float*)(ws + off);    off += (size_t)B_ * C_ * HW_ * 4;   // 32 MB
    _Float16* actv  = (_Float16*)(ws + off); off += (size_t)B_ * HC_ * HW_ * 2;  // 32 MB
    int*      rid   = (int*)(ws + off);      off += (size_t)B_ * HW_ * 4;        // 512 KB
    float*    mu    = (float*)(ws + off);    off += (size_t)B_ * F_ * L_ * 4;    // 80 KB
    float*    Tg    = (float*)(ws + off);    off += (size_t)B_ * F_ * 9 * 64 * 4;
    float*    Tb    = (float*)(ws + off);    off += (size_t)B_ * F_ * 9 * 64 * 4;
    _Float16* frag  = (_Float16*)(ws + off); off += (size_t)2 * NSTEP * 4 * 32 * 16 * 2;

    k_rid     <<<512,   256, 0, stream>>>(segmap, rid);
    k_instnorm<<<B_*C_, 256, 0, stream>>>(x, xnorm);
    k_fc      <<<80,    256, 0, stream>>>(codes, W_fc, b_fc, mu);
    k_regionT <<<180,   256, 0, stream>>>(cgw, cbw, mu, Tg, Tb);
    k_prep    <<<576,   256, 0, stream>>>(sgw, sbw, frag);
    k_conv1   <<<B_*64*H_, 256, 0, stream>>>(mask, ssw, ssb, actv);
    k_main    <<<B_*H_*2,  256, 0, stream>>>(actv, frag, sgb, sbb, rid, Tg, Tb,
                                             cgb, cbb, xnorm, blend_g, blend_b, out);
}
